// VQAttention_32074815767248
// MI455X (gfx1250) — compile-verified, matrix operand order checked by reading, not run on hardware
//
#include <hip/hip_runtime.h>
#include <hip/hip_bf16.h>

typedef __bf16 bf16_t;
typedef __attribute__((ext_vector_type(16))) __bf16 v16bf;
typedef __attribute__((ext_vector_type(8)))  float  v8f;

// Problem shape (fixed by the reference)
constexpr int BB  = 4;
constexpr int LL  = 512;
constexpr int DDim= 1024;
constexpr int HH  = 8;
constexpr int DKK = 128;
constexpr int DVV = 128;
constexpr int SS  = 512;
constexpr int MMc = 512;
constexpr int WWc = 1024; // M + L

#if __has_builtin(__builtin_amdgcn_tensor_load_to_lds) && __has_builtin(__builtin_amdgcn_s_wait_tensorcnt)
#define VQA_USE_TDM 1
typedef unsigned int u32x4 __attribute__((ext_vector_type(4)));
typedef int          i32x4 __attribute__((ext_vector_type(4)));
typedef int          i32x8 __attribute__((ext_vector_type(8)));
#endif

// ---------------- block reductions ----------------
__device__ __forceinline__ float block_reduce_sum(float v, float* red) {
  int t = threadIdx.x, n = blockDim.x;
  red[t] = v; __syncthreads();
  for (int off = n >> 1; off > 0; off >>= 1) {
    if (t < off) red[t] += red[t + off];
    __syncthreads();
  }
  float r = red[0]; __syncthreads();
  return r;
}
__device__ __forceinline__ float block_reduce_max(float v, float* red) {
  int t = threadIdx.x, n = blockDim.x;
  red[t] = v; __syncthreads();
  for (int off = n >> 1; off > 0; off >>= 1) {
    if (t < off) red[t] = fmaxf(red[t], red[t + off]);
    __syncthreads();
  }
  float r = red[0]; __syncthreads();
  return r;
}

#ifdef VQA_USE_TDM
// Issue a TDM 2D tile load: 128 rows x 32 bf16, row stride = strideElems,
// into LDS at byte offset ldsOff with hardware padding 64B->+16B (LDS row
// stride = 80B = 40 bf16, matching the fragment-read layout below).
// D# bitfields per CDNA5 ISA 8.3/8.4 (group0: count/lds/global/type,
// group1: data_size, pad, dims, strides). Groups 2/3 unused (2D tensor).
__device__ __forceinline__ void vqa_tdm_load_tile(const bf16_t* gsrc, unsigned ldsOff,
                                                  int strideElems) {
  unsigned long long ga = (unsigned long long)(size_t)gsrc;
  unsigned ga_lo = __builtin_amdgcn_readfirstlane((unsigned)ga);
  unsigned ga_hi = __builtin_amdgcn_readfirstlane((unsigned)(ga >> 32));
  unsigned lo    = __builtin_amdgcn_readfirstlane(ldsOff);
  int      str   = __builtin_amdgcn_readfirstlane(strideElems);
  u32x4 g0;
  g0[0] = 1u;                                   // count=1, user descriptor
  g0[1] = lo;                                   // lds_addr (bytes)
  g0[2] = ga_lo;                                // global_addr[31:0]
  g0[3] = (ga_hi & 0x01FFFFFFu) | (2u << 30);   // global_addr[56:32] | type=2
  i32x8 g1;
  g1[0] = (1 << 16) | (1 << 20) | (3 << 22) | (3 << 25);
  //        data_size=2B  pad_en   interval=16dw  amount=4dw
  g1[1] = (int)(32u << 16);                     // tensor_dim0 = 32 (low 16 in dw1[31:16])
  g1[2] = (int)(128u << 16);                    // tensor_dim0 hi=0 | tensor_dim1 = 128
  g1[3] = (int)(32u << 16);                     // tensor_dim1 hi=0 | tile_dim0 = 32
  g1[4] = 128;                                  // tile_dim1 = 128, tile_dim2 = 0
  g1[5] = str;                                  // tensor_dim0_stride[31:0]
  g1[6] = 0;                                    // stride hi | dim1_stride lo
  g1[7] = 0;
  i32x4 gz4; gz4[0] = 0; gz4[1] = 0; gz4[2] = 0; gz4[3] = 0;
  i32x8 gz8;
#pragma unroll
  for (int i = 0; i < 8; ++i) gz8[i] = 0;
  __builtin_amdgcn_tensor_load_to_lds(g0, g1, gz4, gz4, gz8, 0);
}
#endif

// ---------------- batched bf16 WMMA GEMM:  C[M,N] = A[M,K] * B[N,K]^T ----------------
// Both operands row-major with K contiguous. ACC=1 accumulates into C.
// Block tile 128x128, K-step 32. 8 waves: 4 M-slots x 2 N-slots, wave tile 32x64.
// TDM path: double-buffered LDS; prefetch tile k+1 via tensor_load_to_lds while
// computing tile k; in-order TENSORcnt lets "wait <= 2" drain the current pair.
template <int ACC>
__global__ __launch_bounds__(256)
void vqa_gemm(const bf16_t* __restrict__ A, const bf16_t* __restrict__ Bm,
              float* __restrict__ C, int Md, int Nd, int Kd,
              long sA, long sB, long sC, int bmod)
{
#ifdef VQA_USE_TDM
  __shared__ __align__(16) bf16_t As[2][128 * 40];  // [buf][row][k], stride 40
  __shared__ __align__(16) bf16_t Bs[2][128 * 40];  // [buf][n][k],   stride 40
#else
  __shared__ __align__(16) bf16_t As[1][128 * 40];
  __shared__ __align__(16) bf16_t Bs[1][128 * 40];
#endif

  const bf16_t* Ab = A  + (long)blockIdx.z * sA;
  const bf16_t* Bb = Bm + (long)(blockIdx.z % bmod) * sB;
  float*        Cb = C  + (long)blockIdx.z * sC;

  const int tid   = threadIdx.x;
  const int lane  = tid & 31;
  const int wid   = tid >> 5;
  const int waveM = (wid & 3) << 5;   // 0..96
  const int waveN = (wid >> 2) << 6;  // 0,64
  const int m0 = blockIdx.y << 7;
  const int n0 = blockIdx.x << 7;
  const int lr   = lane & 15;
  const int half = lane >> 4;

  v8f acc[2][4];
#pragma unroll
  for (int mi = 0; mi < 2; ++mi)
#pragma unroll
    for (int ni = 0; ni < 4; ++ni)
#pragma unroll
      for (int r = 0; r < 8; ++r) acc[mi][ni][r] = 0.0f;

  union Frag { uint4 u[2]; v16bf v; };

#ifdef VQA_USE_TDM
  const unsigned asOff0 = (unsigned)(size_t)(const void*)&As[0][0];
  const unsigned asOff1 = (unsigned)(size_t)(const void*)&As[1][0];
  const unsigned bsOff0 = (unsigned)(size_t)(const void*)&Bs[0][0];
  const unsigned bsOff1 = (unsigned)(size_t)(const void*)&Bs[1][0];
  const int issuer = (__builtin_amdgcn_readfirstlane(wid) == 0);
  const int nk = Kd >> 5;
  int cur = 0;

  if (issuer) {                          // prologue: tile 0 into buffer 0
    vqa_tdm_load_tile(Ab + (size_t)m0 * Kd, asOff0, Kd);
    vqa_tdm_load_tile(Bb + (size_t)n0 * Kd, bsOff0, Kd);
  }
  for (int i = 0; i < nk; ++i) {
    __syncthreads();                     // all waves done reading buf[1-cur] (iter i-1)
    const bool pref = (i + 1 < nk);
    if (pref) {
      if (issuer) {                      // prefetch tile i+1 into the other buffer
        const int k1 = (i + 1) << 5;
        vqa_tdm_load_tile(Ab + (size_t)m0 * Kd + k1, cur ? asOff0 : asOff1, Kd);
        vqa_tdm_load_tile(Bb + (size_t)n0 * Kd + k1, cur ? bsOff0 : bsOff1, Kd);
      }
      __builtin_amdgcn_s_wait_tensorcnt(2);  // tile i done; tile i+1 may be in flight
    } else {
      __builtin_amdgcn_s_wait_tensorcnt(0);
    }
    __syncthreads();                     // tile i visible to all waves

    const bf16_t* Acur = cur ? &As[1][0] : &As[0][0];
    const bf16_t* Bcur = cur ? &Bs[1][0] : &Bs[0][0];
    Frag fa[2], fb[4];
#pragma unroll
    for (int mi = 0; mi < 2; ++mi) {
      const bf16_t* p = Acur + (waveM + mi * 16 + lr) * 40 + half * 8;
      fa[mi].u[0] = *(const uint4*)p;          // K = half*8 + 0..7
      fa[mi].u[1] = *(const uint4*)(p + 16);   // K = 16 + half*8 + 0..7
    }
#pragma unroll
    for (int ni = 0; ni < 4; ++ni) {
      const bf16_t* p = Bcur + (waveN + ni * 16 + lr) * 40 + half * 8;
      fb[ni].u[0] = *(const uint4*)p;
      fb[ni].u[1] = *(const uint4*)(p + 16);
    }
#pragma unroll
    for (int mi = 0; mi < 2; ++mi)
#pragma unroll
      for (int ni = 0; ni < 4; ++ni)
        acc[mi][ni] = __builtin_amdgcn_wmma_f32_16x16x32_bf16(
            false, fa[mi].v, false, fb[ni].v, (short)0, acc[mi][ni], false, false);
    cur ^= 1;
  }
#else
  const int ldrow = tid >> 1, ldseg = tid & 1;  // 128 rows x 32 k tile staging
  for (int k0 = 0; k0 < Kd; k0 += 32) {
    const uint4* ap = (const uint4*)(Ab + (size_t)(m0 + ldrow) * Kd + k0 + ldseg * 16);
    uint4 a0 = ap[0], a1 = ap[1];
    const uint4* bp = (const uint4*)(Bb + (size_t)(n0 + ldrow) * Kd + k0 + ldseg * 16);
    uint4 b0 = bp[0], b1 = bp[1];
    if (k0 + 32 < Kd) {  // prefetch next K tile (global_prefetch_b8)
      __builtin_prefetch(Ab + (size_t)(m0 + ldrow) * Kd + k0 + 32 + ldseg * 16, 0, 1);
      __builtin_prefetch(Bb + (size_t)(n0 + ldrow) * Kd + k0 + 32 + ldseg * 16, 0, 1);
    }
    __syncthreads();
    *(uint4*)&As[0][ldrow * 40 + ldseg * 16]     = a0;
    *(uint4*)&As[0][ldrow * 40 + ldseg * 16 + 8] = a1;
    *(uint4*)&Bs[0][ldrow * 40 + ldseg * 16]     = b0;
    *(uint4*)&Bs[0][ldrow * 40 + ldseg * 16 + 8] = b1;
    __syncthreads();
    Frag fa[2], fb[4];
#pragma unroll
    for (int mi = 0; mi < 2; ++mi) {
      const bf16_t* p = &As[0][(waveM + mi * 16 + lr) * 40 + half * 8];
      fa[mi].u[0] = *(const uint4*)p;
      fa[mi].u[1] = *(const uint4*)(p + 16);
    }
#pragma unroll
    for (int ni = 0; ni < 4; ++ni) {
      const bf16_t* p = &Bs[0][(waveN + ni * 16 + lr) * 40 + half * 8];
      fb[ni].u[0] = *(const uint4*)p;
      fb[ni].u[1] = *(const uint4*)(p + 16);
    }
#pragma unroll
    for (int mi = 0; mi < 2; ++mi)
#pragma unroll
      for (int ni = 0; ni < 4; ++ni)
        acc[mi][ni] = __builtin_amdgcn_wmma_f32_16x16x32_bf16(
            false, fa[mi].v, false, fb[ni].v, (short)0, acc[mi][ni], false, false);
  }
#endif

#pragma unroll
  for (int mi = 0; mi < 2; ++mi)
#pragma unroll
    for (int ni = 0; ni < 4; ++ni) {
      int col = n0 + waveN + ni * 16 + lr;
#pragma unroll
      for (int r = 0; r < 8; ++r) {
        int row = m0 + waveM + mi * 16 + half * 8 + r;
        size_t idx = (size_t)row * Nd + col;
        float v = acc[mi][ni][r];
        Cb[idx] = ACC ? (Cb[idx] + v) : v;
      }
    }
}

// ---------------- elementwise / reduction kernels ----------------
__global__ void vqa_f32_to_bf16(const float* __restrict__ src, bf16_t* __restrict__ dst, long n) {
  long i = blockIdx.x * (long)blockDim.x + threadIdx.x;
  long st = (long)gridDim.x * blockDim.x;
  for (; i < n; i += st) dst[i] = (bf16_t)src[i];
}

// [n1,n2,n3] f32 -> [n1,n3,n2] bf16 (weight / agg transposition to N-major)
__global__ void vqa_t3_to_bf16(const float* __restrict__ src, bf16_t* __restrict__ dst,
                               int n1, int n2, int n3) {
  long n = (long)n1 * n2 * n3;
  long i = blockIdx.x * (long)blockDim.x + threadIdx.x;
  long st = (long)gridDim.x * blockDim.x;
  for (; i < n; i += st) {
    int c = (int)(i % n3);
    long t = i / n3;
    int b = (int)(t % n2);
    int a = (int)(t / n2);
    dst[((size_t)a * n3 + c) * n2 + b] = (bf16_t)src[i];
  }
}

// chunked f32->bf16 copy with independent src/dst chunk strides (k_hat cache concat)
__global__ void vqa_chunk_to_bf16(const float* __restrict__ src, bf16_t* __restrict__ dst,
                                  long chunk, long sstride, long dstride, long nchunks) {
  long n = chunk * nchunks;
  long i = blockIdx.x * (long)blockDim.x + threadIdx.x;
  long st = (long)gridDim.x * blockDim.x;
  for (; i < n; i += st) {
    long c = i / chunk, o = i - c * chunk;
    dst[c * dstride + o] = (bf16_t)src[c * sstride + o];
  }
}

// xl_v_cache [BH,M,dv] f32 -> rvT [BH,dv,W] bf16 (front W<M region)
__global__ void vqa_vcache_T(const float* __restrict__ src, bf16_t* __restrict__ dst) {
  long n = (long)BB * HH * MMc * DVV;
  long i = blockIdx.x * (long)blockDim.x + threadIdx.x;
  long st = (long)gridDim.x * blockDim.x;
  for (; i < n; i += st) {
    int d = (int)(i % DVV);
    long t = i / DVV;
    int w = (int)(t % MMc);
    int bh = (int)(t / MMc);
    dst[((size_t)bh * DVV + d) * WWc + w] = (bf16_t)src[i];
  }
}

// v_raw [B,L,H*dv] f32 -> rvT [BH,dv,M+l] bf16
__global__ void vqa_scatter_vT(const float* __restrict__ vraw, bf16_t* __restrict__ dst) {
  long n = (long)BB * LL * HH * DVV;
  long i = blockIdx.x * (long)blockDim.x + threadIdx.x;
  long st = (long)gridDim.x * blockDim.x;
  for (; i < n; i += st) {
    int d = (int)(i % DVV);
    long t = i / DVV;
    int h = (int)(t % HH); t /= HH;
    int l = (int)(t % LL);
    int b = (int)(t / LL);
    dst[((size_t)(b * HH + h) * DVV + d) * WWc + (MMc + l)] = (bf16_t)vraw[i];
  }
}

// input LayerNorm: xt = LN(x)*g + b  -> bf16 [B*L, D]
__global__ __launch_bounds__(256)
void vqa_ln_x(const float* __restrict__ x, const float* __restrict__ g,
              const float* __restrict__ bvec, bf16_t* __restrict__ xt) {
  __shared__ float red[256];
  const int row = blockIdx.x;
  const float* xr = x + (size_t)row * DDim;
  float vals[4]; float s = 0.f;
#pragma unroll
  for (int j = 0; j < 4; ++j) { vals[j] = xr[threadIdx.x + j * 256]; s += vals[j]; }
  float mu = block_reduce_sum(s, red) * (1.0f / DDim);
  float s2 = 0.f;
#pragma unroll
  for (int j = 0; j < 4; ++j) { float d = vals[j] - mu; s2 += d * d; }
  float var = block_reduce_sum(s2, red) * (1.0f / DDim);
  float rs = rsqrtf(var + 1e-6f);
#pragma unroll
  for (int j = 0; j < 4; ++j) {
    int c = threadIdx.x + j * 256;
    xt[(size_t)row * DDim + c] = (bf16_t)((vals[j] - mu) * rs * g[c] + bvec[c]);
  }
}

// head-wise LN of q and k; q/tau; emit qu=q+u, qvp=q+vp, kln  (bf16, [B*H, L, dk])
__global__ __launch_bounds__(128)
void vqa_headln(const float* __restrict__ qraw, const float* __restrict__ kraw,
                const float* __restrict__ xlu, const float* __restrict__ xlvp,
                bf16_t* __restrict__ qu, bf16_t* __restrict__ qvp, bf16_t* __restrict__ kln) {
  __shared__ float red[128];
  const int bid = blockIdx.x;                 // ((b*L + l)*H + h)
  const int h = bid % HH;
  const int l = (bid / HH) % LL;
  const int b = bid / (HH * LL);
  const int d = threadIdx.x;
  const size_t so   = ((size_t)(b * LL + l)) * (HH * DKK) + (size_t)h * DKK + d;
  const size_t dofs = (((size_t)(b * HH + h)) * LL + l) * DKK + d;
  const float inv_tau = 0.08838834764831845f; // 1/sqrt(128)

  float qv  = qraw[so];
  float mu  = block_reduce_sum(qv, red) * (1.0f / DKK);
  float dq  = qv - mu;
  float var = block_reduce_sum(dq * dq, red) * (1.0f / DKK);
  float qn  = dq * rsqrtf(var + 1e-6f) * inv_tau;
  qu[dofs]  = (bf16_t)(qn + xlu[h * DKK + d]);
  qvp[dofs] = (bf16_t)(qn + xlvp[h * DKK + d]);

  float kv   = kraw[so];
  float mu2  = block_reduce_sum(kv, red) * (1.0f / DKK);
  float dk2  = kv - mu2;
  float var2 = block_reduce_sum(dk2 * dk2, red) * (1.0f / DKK);
  kln[dofs]  = (bf16_t)(dk2 * rsqrtf(var2 + 1e-6f));
}

// codebook: bf16 copy + per-row squared norms
__global__ __launch_bounds__(128)
void vqa_cnorm_cb(const float* __restrict__ codebook, bf16_t* __restrict__ cb,
                  float* __restrict__ cnorm) {
  __shared__ float red[128];
  const int hs = blockIdx.x;
  const float v = codebook[(size_t)hs * DKK + threadIdx.x];
  cb[(size_t)hs * DKK + threadIdx.x] = (bf16_t)v;
  float ss = block_reduce_sum(v * v, red);
  if (threadIdx.x == 0) cnorm[hs] = ss;
}

// VQ argmin over S and straight-through gather k_hat=codebook[z] into recent_k_hat[M+l]
__global__ __launch_bounds__(128)
void vqa_argmin_gather(const float* __restrict__ dots, const float* __restrict__ cnorm,
                       const float* __restrict__ codebook, bf16_t* __restrict__ rkh) {
  __shared__ float rv[128]; __shared__ int ri[128];
  const int bid = blockIdx.x;           // bh*L + l
  const int l  = bid % LL;
  const int bh = bid / LL;
  const int h  = bh % HH;
  const int t  = threadIdx.x;
  float best = 3.0e38f; int bi = 0;
  for (int s = t; s < SS; s += 128) {
    float v = cnorm[h * SS + s] - 2.0f * dots[(size_t)bid * SS + s];
    if (v < best) { best = v; bi = s; }
  }
  rv[t] = best; ri[t] = bi; __syncthreads();
  for (int off = 64; off > 0; off >>= 1) {
    if (t < off) {
      float ov = rv[t + off]; int oi = ri[t + off];
      if (ov < rv[t] || (ov == rv[t] && oi < ri[t])) { rv[t] = ov; ri[t] = oi; }
    }
    __syncthreads();
  }
  const int z = ri[0];
  rkh[(((size_t)bh) * WWc + MMc + l) * DKK + t] =
      (bf16_t)codebook[((size_t)h * SS + z) * DKK + t];
}

// joint softmax over cache(S) + recent(W) with log-count bias, rel-shift gather, causal mask
__global__ __launch_bounds__(256)
void vqa_softmax(float* __restrict__ cs, float* __restrict__ ac,
                 const float* __restrict__ bd, const float* __restrict__ agg_lower,
                 bf16_t* __restrict__ ca, bf16_t* __restrict__ ra) {
  __shared__ float red[256];
  const int bid = blockIdx.x;           // bh*L + l
  const int l  = bid % LL;
  const int bh = bid / LL;
  float* csr = cs + (size_t)bid * SS;
  float* acr = ac + (size_t)bid * WWc;
  const float* bds = bd + (size_t)bh * LL * WWc;
  const float* alr = agg_lower + (size_t)bh * SS;
  const int t = threadIdx.x;

  float mx = -3.0e38f;
  for (int s = t; s < SS; s += 256) {
    float al = alr[s];
    float bias = (al > 0.0f) ? logf(al + 1e-30f) : -1.0e30f;
    float v = csr[s] + bias;
    csr[s] = v;
    mx = fmaxf(mx, v);
  }
  for (int w = t; w < WWc; w += 256) {
    float v;
    if (w <= MMc + l) {
      int tt = LL + l * WWc + w;          // exact Transformer-XL rel_shift index map
      int l2 = tt / (WWc + 1);
      int j2 = tt - l2 * (WWc + 1);
      float bdv = (j2 == 0) ? 0.0f : bds[(size_t)l2 * WWc + (j2 - 1)];
      v = acr[w] + bdv;
    } else v = -1.0e30f;
    acr[w] = v;
    mx = fmaxf(mx, v);
  }
  float m = block_reduce_max(mx, red);
  float sum = 0.0f;
  for (int s = t; s < SS; s += 256)  { float e = __expf(csr[s] - m); csr[s] = e; sum += e; }
  for (int w = t; w < WWc; w += 256) { float e = __expf(acr[w] - m); acr[w] = e; sum += e; }
  float denom = block_reduce_sum(sum, red);
  float rinv = 1.0f / denom;
  for (int s = t; s < SS; s += 256)  ca[(size_t)bid * SS + s]  = (bf16_t)(csr[s] * rinv);
  for (int w = t; w < WWc; w += 256) ra[(size_t)bid * WWc + w] = (bf16_t)(acr[w] * rinv);
}

// y = wv[b,h,l,:] * silu(g_raw[b,l,:])  -> bf16 [B*L, H*dv]
__global__ void vqa_gate(const float* __restrict__ graw, const float* __restrict__ wv,
                         bf16_t* __restrict__ y) {
  long n = (long)BB * LL * HH * DVV;
  long i = blockIdx.x * (long)blockDim.x + threadIdx.x;
  long st = (long)gridDim.x * blockDim.x;
  for (; i < n; i += st) {
    int col = (int)(i % (HH * DVV));
    long rowbl = i / (HH * DVV);
    int h = col / DVV, d = col % DVV;
    int b = (int)(rowbl / LL), l = (int)(rowbl % LL);
    float gv = graw[i];
    float sig = 1.0f / (1.0f + __expf(-gv));
    float wvv = wv[(((size_t)(b * HH + h)) * LL + l) * DVV + d];
    y[i] = (bf16_t)(wvv * gv * sig);
  }
}

// ---------------- host side ----------------
static inline void run_gemm(int ACC, const bf16_t* A, const bf16_t* Bm, float* C,
                            int Md, int Nd, int Kd, long sA, long sB, long sC,
                            int bmod, int batch, hipStream_t s) {
  dim3 g(Nd / 128, Md / 128, batch), b(256, 1, 1);
  if (!ACC) vqa_gemm<0><<<g, b, 0, s>>>(A, Bm, C, Md, Nd, Kd, sA, sB, sC, bmod);
  else      vqa_gemm<1><<<g, b, 0, s>>>(A, Bm, C, Md, Nd, Kd, sA, sB, sC, bmod);
}

extern "C" void kernel_launch(void* const* d_in, const int* in_sizes, int n_in,
                              void* d_out, int out_size, void* d_ws, size_t ws_size,
                              hipStream_t stream) {
  const float* x        = (const float*)d_in[0];
  const float* ln_g     = (const float*)d_in[1];
  const float* ln_b     = (const float*)d_in[2];
  const float* w_q      = (const float*)d_in[3];
  const float* w_k      = (const float*)d_in[4];
  const float* w_v      = (const float*)d_in[5];
  const float* w_g      = (const float*)d_in[6];
  const float* w_res    = (const float*)d_in[7];
  const float* codebook = (const float*)d_in[8];
  const float* xl_u     = (const float*)d_in[9];
  const float* xl_vp    = (const float*)d_in[10];
  const float* xl_r     = (const float*)d_in[11];
  const float* xl_k_hat = (const float*)d_in[12];
  const float* xl_vc    = (const float*)d_in[13];
  const float* agg_up   = (const float*)d_in[14];
  const float* agg_lo   = (const float*)d_in[15];
  float* out = (float*)d_out;

  char* base = (char*)d_ws; size_t off = 0;
  auto alloc = [&](size_t bytes) -> void* {
    void* p = base + off; off = (off + bytes + 255) & ~(size_t)255; return p;
  };
  const size_t BL = (size_t)BB * LL;       // 2048
  const int BH = BB * HH;                  // 32
  // workspace arena (~300 MB total)
  bf16_t* xt    = (bf16_t*)alloc(BL * DDim * 2);
  bf16_t* wqT   = (bf16_t*)alloc((size_t)DDim * DDim * 2);   // [N,K] transposed weights
  bf16_t* wkT   = (bf16_t*)alloc((size_t)DDim * DDim * 2);
  bf16_t* wvT   = (bf16_t*)alloc((size_t)DDim * DDim * 2);
  bf16_t* wgT   = (bf16_t*)alloc((size_t)DDim * DDim * 2);
  bf16_t* wrT   = (bf16_t*)alloc((size_t)DDim * DDim * 2);
  bf16_t* cbB   = (bf16_t*)alloc((size_t)HH * SS * DKK * 2);   // [S,K] (already N-major)
  bf16_t* xlrB  = (bf16_t*)alloc((size_t)HH * WWc * DKK * 2);  // [W,K] (already N-major)
  float*  cnorm = (float*) alloc((size_t)HH * SS * 4);
  float*  qraw  = (float*) alloc(BL * DDim * 4);
  float*  kraw  = (float*) alloc(BL * DDim * 4);
  float*  vraw  = (float*) alloc(BL * DDim * 4);
  float*  graw  = (float*) alloc(BL * DDim * 4);
  bf16_t* quB   = (bf16_t*)alloc((size_t)BH * LL * DKK * 2);
  bf16_t* qvpB  = (bf16_t*)alloc((size_t)BH * LL * DKK * 2);
  bf16_t* klnB  = (bf16_t*)alloc((size_t)BH * LL * DKK * 2);
  float*  dots  = (float*) alloc((size_t)BH * LL * SS * 4);    // reused as cache_scores
  bf16_t* rkhB  = (bf16_t*)alloc((size_t)BH * WWc * DKK * 2);  // [W,K]
  bf16_t* rvT   = (bf16_t*)alloc((size_t)BH * DVV * WWc * 2);  // [dv,W] transposed values
  float*  acF   = (float*) alloc((size_t)BH * LL * WWc * 4);
  float*  bdF   = (float*) alloc((size_t)BH * LL * WWc * 4);
  bf16_t* caB   = (bf16_t*)alloc((size_t)BH * LL * SS * 2);
  bf16_t* raB   = (bf16_t*)alloc((size_t)BH * LL * WWc * 2);
  bf16_t* aggT  = (bf16_t*)alloc((size_t)BH * DVV * SS * 2);   // [dv,S]
  float*  wvO   = (float*) alloc((size_t)BH * LL * DVV * 4);
  bf16_t* yB    = (bf16_t*)alloc(BL * DDim * 2);
  (void)ws_size; (void)n_in; (void)in_sizes; (void)out_size;

  const dim3 cvB(256), cvG(2048);
  // weights -> bf16, transposed to N-major so every GEMM is A*B^T
  vqa_t3_to_bf16<<<cvG, cvB, 0, stream>>>(w_q,   wqT, 1, DDim, DDim);
  vqa_t3_to_bf16<<<cvG, cvB, 0, stream>>>(w_k,   wkT, 1, DDim, DDim);
  vqa_t3_to_bf16<<<cvG, cvB, 0, stream>>>(w_v,   wvT, 1, DDim, DDim);
  vqa_t3_to_bf16<<<cvG, cvB, 0, stream>>>(w_g,   wgT, 1, DDim, DDim);
  vqa_t3_to_bf16<<<cvG, cvB, 0, stream>>>(w_res, wrT, 1, DDim, DDim);
  vqa_t3_to_bf16<<<cvG, cvB, 0, stream>>>(agg_up, aggT, BH, SS, DVV);
  vqa_f32_to_bf16<<<cvG, cvB, 0, stream>>>(xl_r,  xlrB, (long)HH * WWc * DKK);
  // XL caches into the front (w < M) of recent_k_hat / recent_v
  vqa_chunk_to_bf16<<<cvG, cvB, 0, stream>>>(xl_k_hat, rkhB, (long)MMc * DKK,
                                             (long)MMc * DKK, (long)WWc * DKK, BH);
  vqa_vcache_T<<<cvG, cvB, 0, stream>>>(xl_vc, rvT);
  vqa_cnorm_cb<<<HH * SS, 128, 0, stream>>>(codebook, cbB, cnorm);

  // input LN, projections
  vqa_ln_x<<<(int)BL, 256, 0, stream>>>(x, ln_g, ln_b, xt);
  run_gemm(0, xt, wqT, qraw, (int)BL, DDim, DDim, 0, 0, 0, 1, 1, stream);
  run_gemm(0, xt, wkT, kraw, (int)BL, DDim, DDim, 0, 0, 0, 1, 1, stream);
  run_gemm(0, xt, wvT, vraw, (int)BL, DDim, DDim, 0, 0, 0, 1, 1, stream);
  run_gemm(0, xt, wgT, graw, (int)BL, DDim, DDim, 0, 0, 0, 1, 1, stream);

  // head LN, VQ
  vqa_headln<<<BB * LL * HH, 128, 0, stream>>>(qraw, kraw, xl_u, xl_vp, quB, qvpB, klnB);
  run_gemm(0, klnB, cbB, dots, LL, SS, DKK,
           (long)LL * DKK, (long)SS * DKK, (long)LL * SS, HH, BH, stream);
  vqa_argmin_gather<<<BH * LL, 128, 0, stream>>>(dots, cnorm, codebook, rkhB);
  vqa_scatter_vT<<<cvG, cvB, 0, stream>>>(vraw, rvT);

  // scores: cache (reuse dots buffer), ac, bd
  run_gemm(0, quB, cbB, dots, LL, SS, DKK,
           (long)LL * DKK, (long)SS * DKK, (long)LL * SS, HH, BH, stream);
  run_gemm(0, quB, rkhB, acF, LL, WWc, DKK,
           (long)LL * DKK, (long)WWc * DKK, (long)LL * WWc, BH, BH, stream);
  run_gemm(0, qvpB, xlrB, bdF, LL, WWc, DKK,
           (long)LL * DKK, (long)WWc * DKK, (long)LL * WWc, HH, BH, stream);

  // joint softmax -> bf16 probabilities
  vqa_softmax<<<BH * LL, 256, 0, stream>>>(dots, acF, bdF, agg_lo, caB, raB);

  // wv = ra @ recent_v + ca @ agg_upper   (values pre-transposed to [dv, K])
  run_gemm(0, raB, rvT, wvO, LL, DVV, WWc,
           (long)LL * WWc, (long)DVV * WWc, (long)LL * DVV, BH, BH, stream);
  run_gemm(1, caB, aggT, wvO, LL, DVV, SS,
           (long)LL * SS, (long)DVV * SS, (long)LL * DVV, BH, BH, stream);

  // gate + output projection
  vqa_gate<<<cvG, cvB, 0, stream>>>(graw, wvO, yB);
  run_gemm(0, yB, wrT, out, (int)BL, DDim, DDim, 0, 0, 0, 1, 1, stream);
}
